// Graph_Decoder_Norm_32212254720651
// MI455X (gfx1250) — matrix operand order repaired and checked by reading, not compiled, not run on hardware
//
#include <hip/hip_runtime.h>
#include <hip/hip_bf16.h>

typedef __attribute__((ext_vector_type(16))) _Float16 v16h;
typedef __attribute__((ext_vector_type(8)))  _Float16 v8h;
typedef __attribute__((ext_vector_type(4)))  _Float16 v4h;
typedef __attribute__((ext_vector_type(8)))  float    v8f;

#define DREF 128

union AU { v16h v; struct { v8h lo; v8h hi; } s; };

// ---------------- degree / dinv ----------------
__global__ void k_deg_init(float* deg, int n) {
    int i = blockIdx.x * blockDim.x + threadIdx.x;
    if (i < n) deg[i] = 1.0f;   // self loop
}

__global__ void k_deg_count(const int* __restrict__ dst, float* deg, int e) {
    int i = blockIdx.x * blockDim.x + threadIdx.x;
    if (i < e)
        __hip_atomic_fetch_add(&deg[dst[i]], 1.0f, __ATOMIC_RELAXED, __HIP_MEMORY_SCOPE_AGENT);
}

__global__ void k_dinv(float* deg, int n) {
    int i = blockIdx.x * blockDim.x + threadIdx.x;
    if (i < n) deg[i] = rsqrtf(fmaxf(deg[i], 1.0f));
}

// ---------------- f32 -> f16 convert (activations) ----------------
__global__ void k_f32_to_f16(const float* __restrict__ in, _Float16* __restrict__ out, int n) {
    int i = blockIdx.x * blockDim.x + threadIdx.x;
    if (i < n) out[i] = (_Float16)in[i];
}

// ---------------- W pack: f32 row-major [K][N] -> f16 WMMA-B fragment layout --
// Bp[((kb*8+tn)*32 + lane)*16 + j] = W[kb*32 + 16*(lane>>4) + j][tn*16 + (lane&15)]
__global__ void k_pack_W(const float* __restrict__ W, _Float16* __restrict__ Bp) {
    int i = blockIdx.x * blockDim.x + threadIdx.x;   // 0 .. 128*128-1
    if (i >= DREF * DREF) return;
    int j    = i & 15;
    int lane = (i >> 4) & 31;
    int tn   = (i >> 9) & 7;
    int kb   = (i >> 12) & 3;
    int k = kb * 32 + ((lane >> 4) << 4) + j;
    int n = tn * 16 + (lane & 15);
    Bp[i] = (_Float16)W[k * DREF + n];
}

// ---------------- WMMA GEMM: C[M,128] = A[M,128] * W[128,128] ----------------
// One wave per (column-tile tn); B fragments persistent in registers,
// wave grid-strides over m-tiles streaming A and C.
__global__ __launch_bounds__(32)
void k_gemm_wmma(const _Float16* __restrict__ A, const _Float16* __restrict__ Bp,
                 float* __restrict__ C, int Mtiles) {
    const int lane    = threadIdx.x & 31;
    const int half_id = lane >> 4;
    const int lid     = lane & 15;
    const int tn      = blockIdx.y;

    // Load all 4 K-block B fragments once (contiguous 32B per lane).
    v16h b[4];
#pragma unroll
    for (int kb = 0; kb < 4; ++kb)
        b[kb] = *(const v16h*)(Bp + (size_t)(((kb * 8 + tn) * 32) + lane) * 16);

    for (int tm = blockIdx.x; tm < Mtiles; tm += gridDim.x) {
        // Single A base pointer; K-block offsets become load immediates.
        const _Float16* ap = A + ((size_t)tm * 16 + lid) * DREF + half_id * 16;
        v8f c = {};
#pragma unroll
        for (int kb = 0; kb < 4; ++kb) {
            AU au;
            au.s.lo = *(const v8h*)(ap + kb * 32);
            au.s.hi = *(const v8h*)(ap + kb * 32 + 8);
            c = __builtin_amdgcn_wmma_f32_16x16x32_f16(false, au.v, false, b[kb],
                                                       (short)0, c, false, false);
        }
        // Single C base pointer; row offsets (j*512B) become store immediates.
        float* cp = C + ((size_t)tm * 16 + 8 * half_id) * DREF + tn * 16 + lid;
#pragma unroll
        for (int j = 0; j < 8; ++j)
            cp[j * DREF] = c[j];
    }
}

// ---------------- aggregation ----------------
// acc = bias + self-loop contribution (xw * dinv^2)
__global__ void k_agg_init(const float* __restrict__ xw, const float* __restrict__ dinv,
                           const float* __restrict__ bias, float* __restrict__ acc, int n) {
    int i = blockIdx.x * blockDim.x + threadIdx.x;
    if (i < n * DREF) {
        int node = i >> 7, d = i & 127;
        float di = dinv[node];
        acc[i] = bias[d] + xw[i] * di * di;
    }
}

// one thread = (edge, 4-feature chunk); float4 gather + 4 f32 atomic adds
__global__ __launch_bounds__(256)
void k_agg_edges(const int* __restrict__ src, const int* __restrict__ dst,
                 const float* __restrict__ xw, const float* __restrict__ dinv,
                 float* __restrict__ acc, int E) {
    long long tid   = (long long)blockIdx.x * blockDim.x + threadIdx.x;
    long long total = (long long)E * 32;
    long long step  = (long long)gridDim.x * blockDim.x;
    for (; tid < total; tid += step) {
        int e = (int)(tid >> 5);
        int c = (int)(tid & 31) * 4;
        int s = src[e], d = dst[e];
        float w = dinv[s] * dinv[d];
        const float4 v = *(const float4*)(xw + (size_t)s * DREF + c);
        float* out = acc + (size_t)d * DREF + c;
        __hip_atomic_fetch_add(out + 0, v.x * w, __ATOMIC_RELAXED, __HIP_MEMORY_SCOPE_AGENT);
        __hip_atomic_fetch_add(out + 1, v.y * w, __ATOMIC_RELAXED, __HIP_MEMORY_SCOPE_AGENT);
        __hip_atomic_fetch_add(out + 2, v.z * w, __ATOMIC_RELAXED, __HIP_MEMORY_SCOPE_AGENT);
        __hip_atomic_fetch_add(out + 3, v.w * w, __ATOMIC_RELAXED, __HIP_MEMORY_SCOPE_AGENT);
    }
}

// ---------------- LayerNorm + GELU (one wave32 per node) ----------------
__device__ __forceinline__ float gelu_tanh(float x) {
    const float k0 = 0.7978845608028654f;  // sqrt(2/pi)
    float t = tanhf(k0 * (x + 0.044715f * x * x * x));
    return 0.5f * x * (1.0f + t);
}

__global__ __launch_bounds__(256)
void k_ln_gelu(const float* __restrict__ acc, const float* __restrict__ gamma,
               const float* __restrict__ beta, float* __restrict__ xout,
               _Float16* __restrict__ xh, int N) {
    int wave = (int)((blockIdx.x * (long long)blockDim.x + threadIdx.x) >> 5);
    int lane = threadIdx.x & 31;
    if (wave >= N) return;

    const float4 v = *(const float4*)(acc + (size_t)wave * DREF + lane * 4);
    float s  = v.x + v.y + v.z + v.w;
    float s2 = v.x * v.x + v.y * v.y + v.z * v.z + v.w * v.w;
#pragma unroll
    for (int off = 16; off; off >>= 1) {
        s  += __shfl_xor(s,  off, 32);
        s2 += __shfl_xor(s2, off, 32);
    }
    float mu   = s * (1.0f / DREF);
    float var  = s2 * (1.0f / DREF) - mu * mu;
    float rstd = rsqrtf(var + 1e-5f);

    float o[4];
    float in[4] = {v.x, v.y, v.z, v.w};
#pragma unroll
    for (int j = 0; j < 4; ++j) {
        int d = lane * 4 + j;
        float y = (in[j] - mu) * rstd * gamma[d] + beta[d];
        o[j] = gelu_tanh(y);
    }
    float4 of = {o[0], o[1], o[2], o[3]};
    *(float4*)(xout + (size_t)wave * DREF + lane * 4) = of;
    v4h oh = {(_Float16)o[0], (_Float16)o[1], (_Float16)o[2], (_Float16)o[3]};
    *(v4h*)(xh + (size_t)wave * DREF + lane * 4) = oh;
}

// ---------------- driver ----------------
extern "C" void kernel_launch(void* const* d_in, const int* in_sizes, int n_in,
                              void* d_out, int out_size, void* d_ws, size_t ws_size,
                              hipStream_t stream) {
    const float* z      = (const float*)d_in[0];
    const int*   ei     = (const int*)d_in[1];
    const float* Ws     = (const float*)d_in[2];
    const float* bs     = (const float*)d_in[3];
    const float* gammas = (const float*)d_in[4];
    const float* betas  = (const float*)d_in[5];

    const int N = in_sizes[0] / DREF;
    const int E = in_sizes[1] / 2;
    const int L = in_sizes[2] / (DREF * DREF);
    const int* src = ei;
    const int* dst = ei + E;

    auto up = [](size_t x) { return (x + 255) & ~(size_t)255; };
    char* p = (char*)d_ws;
    float*    dinv = (float*)p;     p += up((size_t)N * sizeof(float));
    float*    acc  = (float*)p;     p += up((size_t)N * DREF * sizeof(float));
    float*    xw   = (float*)p;     p += up((size_t)N * DREF * sizeof(float));
    _Float16* xh   = (_Float16*)p;  p += up((size_t)N * DREF * sizeof(_Float16));
    _Float16* Bp   = (_Float16*)p;  p += up((size_t)DREF * DREF * sizeof(_Float16));

    const int T = 256;
    // degrees -> dinv
    k_deg_init<<<(N + T - 1) / T, T, 0, stream>>>(dinv, N);
    k_deg_count<<<(E + T - 1) / T, T, 0, stream>>>(dst, dinv, E);
    k_dinv<<<(N + T - 1) / T, T, 0, stream>>>(dinv, N);

    // f16 copy of input features
    k_f32_to_f16<<<(N * DREF + T - 1) / T, T, 0, stream>>>(z, xh, N * DREF);

    const int Mtiles = N / 16;                    // 625
    dim3 ggrid(125, DREF / 16);                   // 1000 waves, 5 m-tiles each
    long long ework = (long long)E * 32;
    int eblocks = (int)((ework + T - 1) / T);

    for (int i = 0; i < L; ++i) {
        // weights -> packed f16 B fragments
        k_pack_W<<<(DREF * DREF + T - 1) / T, T, 0, stream>>>(
            Ws + (size_t)i * DREF * DREF, Bp);
        // xw = x @ W  (WMMA, B persistent in registers)
        k_gemm_wmma<<<ggrid, 32, 0, stream>>>(xh, Bp, xw, Mtiles);
        // acc = bias + self-loop
        k_agg_init<<<(N * DREF + T - 1) / T, T, 0, stream>>>(xw, dinv, bs + (size_t)i * DREF, acc, N);
        // scatter-add over edges
        k_agg_edges<<<eblocks, T, 0, stream>>>(src, dst, xw, dinv, acc, E);
        // layernorm + gelu -> next x (f32 into xw or final d_out) + f16 copy
        float* xo = (i == L - 1) ? (float*)d_out : xw;
        k_ln_gelu<<<(N * 32 + T - 1) / T, T, 0, stream>>>(
            acc, gammas + (size_t)i * DREF, betas + (size_t)i * DREF, xo, xh, N);
    }
}
